// Message_19250043421112
// MI455X (gfx1250) — matrix operand order repaired and checked by reading, not compiled, not run on hardware
//
#include <hip/hip_runtime.h>
#include <cstdint>
#include <cstddef>

#define N_FEAT 64
#define HID    256
#define B_DIM  4
#define EPS_C  0.1f
#define R0_C   10.0f

typedef __attribute__((ext_vector_type(16))) __bf16 v16bf;
typedef __attribute__((ext_vector_type(8)))  float  v8f;

struct Bits256 { uint4 lo, hi; };
struct Acc4 { v8f t0, t1, t2, t3; };

// ---------- small helpers ----------
__device__ __forceinline__ unsigned short f2bf(float f) {
  unsigned u = __builtin_bit_cast(unsigned, f);
  unsigned r = u + 0x7FFFu + ((u >> 16) & 1u);   // round-to-nearest-even
  return (unsigned short)(r >> 16);
}
__device__ __forceinline__ float silu_fast(float v) {
  // v * sigmoid(v) with HW v_rcp_f32 instead of IEEE divide
  return v * __builtin_amdgcn_rcpf(1.0f + __expf(-v));
}
__device__ __forceinline__ void lds_fence() {
  asm volatile("s_wait_dscnt 0" ::: "memory");
}

// ---------- WMMA fragment loaders ----------
// B-fragment memory layout (produced by pack_weights): per 16x16x32 K-slice,
// a 512-half block; element [lane*16 + h] = B[k][n] with
//   k = kbase + (lane>>4)*16 + h,  n = nbase + (lane&15)
__device__ __forceinline__ v16bf load_b_frag(const unsigned short* blk, int lane) {
  Bits256 t;
  const uint4* p = (const uint4*)(blk + lane * 16);
  t.lo = p[0];
  t.hi = p[1];
  return __builtin_bit_cast(v16bf, t);
}

// A-fragment from LDS rows (bf16, row-major, padded pitch). 16-bit A 16x32 in this code's contiguous k order:
// lane half=lane>>4; halves 0..15 hold k = kbase + half*16 + 0..15
// (matches B-fragment packing) — two 16B LDS loads.
__device__ __forceinline__ v16bf load_a_frag(const unsigned short* rows, int pitch,
                                             int kt, int lane) {
  int m = lane & 15, half = lane >> 4;
  const unsigned short* p = rows + m * pitch + kt * 32 + half * 16;
  Bits256 t;
  t.lo = *(const uint4*)p;
  t.hi = *(const uint4*)(p + 8);
  return __builtin_bit_cast(v16bf, t);
}

// 4 output tiles computed concurrently: one A-fragment per K-step feeds four
// independent WMMAs -> no D->C RAW back-to-back stalls, 4x less LDS A traffic.
// Accumulators start from the inline-0 SRC2 form (v8f{} folds to literal 0);
// bias is applied in the caller's epilogue.
template<int KT>
__device__ __forceinline__ Acc4 mma_tile4(const unsigned short* rows, int pitch,
                                          const unsigned short* wfrag,
                                          int nt0, int ntStride, int lane) {
  Acc4 s;
  s.t0 = {};
  s.t1 = {};
  s.t2 = {};
  s.t3 = {};
  const unsigned short* blk0 = wfrag + ((nt0               ) * KT) * 512;
  const unsigned short* blk1 = wfrag + ((nt0 +     ntStride) * KT) * 512;
  const unsigned short* blk2 = wfrag + ((nt0 + 2 * ntStride) * KT) * 512;
  const unsigned short* blk3 = wfrag + ((nt0 + 3 * ntStride) * KT) * 512;
#pragma unroll
  for (int kt = 0; kt < KT; ++kt) {
    v16bf a  = load_a_frag(rows, pitch, kt, lane);
    v16bf w0 = load_b_frag(blk0 + kt * 512, lane);
    v16bf w1 = load_b_frag(blk1 + kt * 512, lane);
    v16bf w2 = load_b_frag(blk2 + kt * 512, lane);
    v16bf w3 = load_b_frag(blk3 + kt * 512, lane);
    s.t0 = __builtin_amdgcn_wmma_f32_16x16x32_bf16(false, a, false, w0, (short)0, s.t0, false, false);
    s.t1 = __builtin_amdgcn_wmma_f32_16x16x32_bf16(false, a, false, w1, (short)0, s.t1, false, false);
    s.t2 = __builtin_amdgcn_wmma_f32_16x16x32_bf16(false, a, false, w2, (short)0, s.t2, false, false);
    s.t3 = __builtin_amdgcn_wmma_f32_16x16x32_bf16(false, a, false, w3, (short)0, s.t3, false, false);
  }
  return s;
}

// One MLP layer: 16 edges x (KT*32 -> 256). Bias + optional SiLU in epilogue.
// F32OUT=false: bf16 result rows (next WMMA A operand). F32OUT=true: f32 rows.
// C/D layout: lane n = lane&15, rows m = r + 8*(lane>>4).
template<int KT, bool SILU, bool F32OUT>
__device__ __forceinline__ void mlp_layer(const unsigned short* rows, int apitch,
                                          const unsigned short* wfrag,
                                          const float* __restrict__ bias,
                                          void* dstv, int dpitch, int lane) {
  const int n = lane & 15, hm = lane >> 4;
  unsigned short* dsth = (unsigned short*)dstv;
  float*          dstf = (float*)dstv;
  for (int ntg = 0; ntg < 4; ++ntg) {
    const int nt0 = ntg * 4;
    Acc4 s = mma_tile4<KT>(rows, apitch, wfrag, nt0, 1, lane);
    const v8f* accs[4] = {&s.t0, &s.t1, &s.t2, &s.t3};
#pragma unroll
    for (int i = 0; i < 4; ++i) {
      const float bv = bias[(nt0 + i) * 16 + n];
#pragma unroll
      for (int r = 0; r < 8; ++r) {
        float v = (*accs[i])[r] + bv;
        if (SILU) v = silu_fast(v);
        const int off = (r + 8 * hm) * dpitch + (nt0 + i) * 16 + n;
        if (F32OUT) dstf[off] = v;
        else        dsth[off] = f2bf(v);
      }
    }
  }
  lds_fence();
}

// ---------- per-wave LDS layout (bytes), padded pitches vs 64-bank LDS ----------
#define XP 72          // Xin pitch (halves)
#define HPP 264        // hidden pitch (halves)
#define DPP 256        // dist-out pitch (floats)
#define OFF_XIN  0                        // 16*72*2  = 2304
#define OFF_HP   2304                     // 16*264*2 = 8448
#define OFF_HQ   (2304 + 8448)
#define OFF_DOUT (2304 + 8448 + 8448)     // 16*256*4 = 16384 (f32)
#define OFF_MISC (OFF_DOUT + 16384)       // dir 192 + dist 64 + src 64 + dst 64
#define WAVE_LDS 35968                    // 8 waves -> 287744 B < 320 KB

// frag buffer offsets in halves (per-W sizes: W1 16384, W2/W3 65536 each)
#define FD1 0
#define FD2 16384
#define FD3 81920
#define FA1 147456
#define FA2 163840
#define FA3 229376
#define FRAG_TOTAL 294912

// ---------- weight pre-pack: fp32 row-major -> bf16 WMMA B-fragments ----------
__global__ __launch_bounds__(256) void pack_weights(
    const float* __restrict__ dW1, const float* __restrict__ dW2, const float* __restrict__ dW3,
    const float* __restrict__ aW1, const float* __restrict__ aW2, const float* __restrict__ aW3,
    unsigned short* __restrict__ frag) {
  int idx = blockIdx.x * blockDim.x + threadIdx.x;
  if (idx >= FRAG_TOTAL) return;
  const float* W; int KT; int rel;
  if      (idx < FD2) { W = dW1; KT = 2; rel = idx - FD1; }
  else if (idx < FD3) { W = dW2; KT = 8; rel = idx - FD2; }
  else if (idx < FA1) { W = dW3; KT = 8; rel = idx - FD3; }
  else if (idx < FA2) { W = aW1; KT = 2; rel = idx - FA1; }
  else if (idx < FA3) { W = aW2; KT = 8; rel = idx - FA2; }
  else                { W = aW3; KT = 8; rel = idx - FA3; }
  int block = rel >> 9;
  int r     = rel & 511;
  int lane  = r >> 4;
  int h     = r & 15;
  int nt = block / KT;
  int kt = block % KT;
  int k = kt * 32 + ((lane >> 4) << 4) + h;
  int n = nt * 16 + (lane & 15);
  frag[idx] = f2bf(W[k * HID + n]);
}

__global__ __launch_bounds__(256) void zero_f32(float* __restrict__ p, int n) {
  int i = blockIdx.x * 256 + threadIdx.x;
  if (i < n) p[i] = 0.0f;
}

// ---------- fused message-passing kernel ----------
__global__ __launch_bounds__(256) void fused_mp(
    const float* __restrict__ x_a, const float* __restrict__ x_v,
    const float* __restrict__ r_ij,
    const int* __restrict__ src, const int* __restrict__ dst,
    const float* __restrict__ db1, const float* __restrict__ db2, const float* __restrict__ db3,
    const float* __restrict__ ab1, const float* __restrict__ ab2, const float* __restrict__ ab3,
    const unsigned short* __restrict__ frag,
    float* __restrict__ out_a, float* __restrict__ out_v,
    int A, int E, int nTiles) {
  extern __shared__ char smem[];
  const int wave = threadIdx.x >> 5;
  const int lane = threadIdx.x & 31;
  char* wb = smem + wave * WAVE_LDS;
  unsigned short* Xin  = (unsigned short*)(wb + OFF_XIN);
  unsigned short* Hp   = (unsigned short*)(wb + OFF_HP);
  unsigned short* Hq   = (unsigned short*)(wb + OFF_HQ);
  float*          DoutF = (float*)(wb + OFF_DOUT);
  float* dirL  = (float*)(wb + OFF_MISC);
  float* distL = (float*)(wb + OFF_MISC + 192);
  int*   srcL  = (int*)(wb + OFF_MISC + 256);
  int*   dstL  = (int*)(wb + OFF_MISC + 320);

  const unsigned short* fragD1 = frag + FD1;
  const unsigned short* fragD2 = frag + FD2;
  const unsigned short* fragD3 = frag + FD3;
  const unsigned short* fragA1 = frag + FA1;
  const unsigned short* fragA2 = frag + FA2;
  const unsigned short* fragA3 = frag + FA3;

  const int tile = blockIdx.x * 8 + wave;   // wave-uniform
  if (tile >= nTiles) return;
  const int e0 = tile * 16;

  __builtin_prefetch(fragA2, 0, 3);         // weights stay hot in L2
  __builtin_prefetch(fragA3, 0, 3);

  // ---- stage per-edge scalars (lanes 0..15 each own one edge) ----
  if (lane < 16) {
    int e = e0 + lane;
    if (e < E) {
      float rx = r_ij[e * 3 + 0], ry = r_ij[e * 3 + 1], rz = r_ij[e * 3 + 2];
      float d2 = rx * rx + ry * ry + rz * rz;
      float inv = rsqrtf(EPS_C + d2);
      dirL[lane * 3 + 0] = rx * inv;
      dirL[lane * 3 + 1] = ry * inv;
      dirL[lane * 3 + 2] = rz * inv;
      distL[lane] = sqrtf(d2);
      srcL[lane] = src[e];
      dstL[lane] = dst[e];
    } else {
      dirL[lane * 3 + 0] = 0.f; dirL[lane * 3 + 1] = 0.f; dirL[lane * 3 + 2] = 0.f;
      distL[lane] = 0.f; srcL[lane] = -1; dstL[lane] = 0;
    }
  }
  lds_fence();

  // ---- radial encoding -> Xin (bf16) ----
  {
    int m = lane & 15, cb = (lane >> 4) * 32;
    float dist = distL[m];
    const float k0 = 3.14159265358979323846f / R0_C;
#pragma unroll
    for (int j = 0; j < 32; ++j) {
      int c = cb + j;
      int q = (c < 32) ? c : (c - 32);
      float ph = k0 * (float)(1 + (q >> 1)) * dist;
      float v = (c < 32) ? __cosf(ph) : __sinf(ph);
      Xin[m * XP + c] = f2bf(v);
    }
  }
  lds_fence();

  // ---- distance MLP (edge-only, shared over B) -> DoutF (f32) ----
  mlp_layer<2, true , false>(Xin, XP,  fragD1, db1, Hp,    HPP, lane);
  mlp_layer<8, true , false>(Hp,  HPP, fragD2, db2, Hq,    HPP, lane);
  mlp_layer<8, false, true >(Hq,  HPP, fragD3, db3, DoutF, DPP, lane);

  const int nn = lane & 15, hm = lane >> 4;

  for (int b = 0; b < B_DIM; ++b) {
    // ---- gather x_a[b, dst, :] -> Xin (bf16) ----
    {
      int m = lane & 15, cb = (lane >> 4) * 32;
      int ad = (dstL[m] < 0) ? 0 : dstL[m];
      const float* xa = x_a + (size_t)(b * A + ad) * N_FEAT + cb;
#pragma unroll
      for (int j = 0; j < 32; j += 4) {
        float4 f = *(const float4*)(xa + j);
        Xin[m * XP + cb + j + 0] = f2bf(f.x);
        Xin[m * XP + cb + j + 1] = f2bf(f.y);
        Xin[m * XP + cb + j + 2] = f2bf(f.z);
        Xin[m * XP + cb + j + 3] = f2bf(f.w);
      }
    }
    lds_fence();

    mlp_layer<2, true, false>(Xin, XP,  fragA1, ab1, Hp, HPP, lane);
    mlp_layer<8, true, false>(Hp,  HPP, fragA2, ab2, Hq, HPP, lane);

    // ---- final layer in 4 channel-groups: tiles {g, g+4, g+8, g+12} give
    //      prod[:,c], prod[:,c+64], prod[:,c+128], prod[:,c+192] at the same
    //      (lane, row) positions -> shuffle-free epilogue ----
    for (int g = 0; g < 4; ++g) {
      Acc4 s = mma_tile4<8>(Hq, HPP, fragA3, g, 4, lane);
      const float bb0 = ab3[(g     ) * 16 + nn];
      const float bb1 = ab3[(g +  4) * 16 + nn];
      const float bb2 = ab3[(g +  8) * 16 + nn];
      const float bb3 = ab3[(g + 12) * 16 + nn];
#pragma unroll
      for (int r = 0; r < 8; ++r) {
        int m = r + 8 * hm;
        int aS = srcL[m];
        if (aS < 0) continue;
        int c = g * 16 + nn;
        float p0 = (s.t0[r] + bb0) * DoutF[m * DPP + c      ];
        float p1 = (s.t1[r] + bb1) * DoutF[m * DPP + c +  64];
        float p2 = (s.t2[r] + bb2) * DoutF[m * DPP + c + 128];
        float p3 = (s.t3[r] + bb3) * DoutF[m * DPP + c + 192];
        int aD = dstL[m];
        const float* xv = x_v + (size_t)((b * A + aD) * N_FEAT + c) * 3;
        float vx = xv[0], vy = xv[1], vz = xv[2];
        float dx = dirL[m * 3 + 0], dy = dirL[m * 3 + 1], dz = dirL[m * 3 + 2];
        float cx = vy * dz - vz * dy;
        float cy = vz * dx - vx * dz;
        float cz = vx * dy - vy * dx;
        float ox = vx * p0 + cx * p1 + dx * p2;
        float oy = vy * p0 + cy * p1 + dy * p2;
        float oz = vz * p0 + cz * p1 + dz * p2;
        float* ov = out_v + (size_t)((b * A + aS) * N_FEAT + c) * 3;
        unsafeAtomicAdd(ov + 0, ox);
        unsafeAtomicAdd(ov + 1, oy);
        unsafeAtomicAdd(ov + 2, oz);
        unsafeAtomicAdd(out_a + (size_t)(b * A + aS) * N_FEAT + c, p3);
      }
    }
  }
}

// ---------- host entry ----------
extern "C" void kernel_launch(void* const* d_in, const int* in_sizes, int n_in,
                              void* d_out, int out_size, void* d_ws, size_t ws_size,
                              hipStream_t stream) {
  const float* x_a  = (const float*)d_in[0];
  const float* x_v  = (const float*)d_in[1];
  const float* r_ij = (const float*)d_in[2];
  const int*   src  = (const int*)d_in[3];
  const int*   dst  = (const int*)d_in[4];
  const float* dW1 = (const float*)d_in[5];
  const float* db1 = (const float*)d_in[6];
  const float* dW2 = (const float*)d_in[7];
  const float* db2 = (const float*)d_in[8];
  const float* dW3 = (const float*)d_in[9];
  const float* db3 = (const float*)d_in[10];
  const float* aW1 = (const float*)d_in[11];
  const float* ab1 = (const float*)d_in[12];
  const float* aW2 = (const float*)d_in[13];
  const float* ab2 = (const float*)d_in[14];
  const float* aW3 = (const float*)d_in[15];
  const float* ab3 = (const float*)d_in[16];

  const int E = in_sizes[3];
  const int A = in_sizes[0] / (B_DIM * N_FEAT);

  unsigned short* frag = (unsigned short*)d_ws;  // needs ~590 KB of scratch

  zero_f32<<<(out_size + 255) / 256, 256, 0, stream>>>((float*)d_out, out_size);
  pack_weights<<<(FRAG_TOTAL + 255) / 256, 256, 0, stream>>>(dW1, dW2, dW3,
                                                             aW1, aW2, aW3, frag);

  const int nTiles = (E + 15) / 16;
  const int wgs = (nTiles + 7) / 8;
  float* out_a = (float*)d_out;
  float* out_v = out_a + (size_t)B_DIM * A * N_FEAT;
  fused_mp<<<wgs, 256, 8 * WAVE_LDS, stream>>>(x_a, x_v, r_ij, src, dst,
                                               db1, db2, db3, ab1, ab2, ab3,
                                               frag, out_a, out_v, A, E, nTiles);
}
